// DynamicGCNModel_47330539602430
// MI455X (gfx1250) — matrix operand order repaired and checked by reading, not compiled, not run on hardware
//
#include <hip/hip_runtime.h>
#include <hip/hip_bf16.h>

// ---------------------------------------------------------------------------
// DynamicGCN pipeline for gfx1250 (MI455X, wave32, WMMA)
//   N=50000 nodes, E=1.6M edges, C=128 channels
//   Dense GEMMs -> v_wmma_f32_16x16x32_bf16 (f32 accumulate)
//   Graph aggregation -> coalesced gather + f32 L2 atomics
// ---------------------------------------------------------------------------

#define CCH   128          // channels
#define BNEPS 1e-5f

typedef __attribute__((ext_vector_type(16))) __bf16 v16bf;
typedef __attribute__((ext_vector_type(8)))  float  v8f;

// ---------------------------------------------------------------------------
// GEMM: out[nRows, COUT] = A[nRows, CIN] @ W[COUT, CIN]^T (+ bias)
// One wave per 16-row tile (nRows % 16 == 0; 50000 = 3125*16).
// A fragments (16x32 bf16, ISA 7.12.2 layout) cached in registers across the
// full K dimension and reused for every 16-wide N tile.
// ---------------------------------------------------------------------------
template <int CIN, int COUT>
__global__ __launch_bounds__(128) void gemm_wmma_kernel(
    float* __restrict__ out, const float* __restrict__ A,
    const float* __restrict__ W, const float* __restrict__ bias, int nRows) {
  const int lane = threadIdx.x & 31;
  const int wave = threadIdx.x >> 5;
  const int tile = blockIdx.x * 4 + wave;         // 16-row tile index
  const int nTiles = nRows >> 4;
  if (tile >= nTiles) return;

  const int row0 = tile << 4;
  const int m = lane & 15;                        // row within tile / B column
  const int hh = lane >> 4;                       // lane half

  // --- load A fragments once (convert f32 -> bf16 in-register) -------------
  v16bf afr[CIN / 32];
  const float* arow = A + (size_t)(row0 + m) * CIN;
#pragma unroll
  for (int kt = 0; kt < CIN / 32; ++kt) {
    const float* p0 = arow + kt * 32 + hh * 8;    // K chunk [h*8, h*8+8)
    const float* p1 = p0 + 16;                    // K chunk [16+h*8, ...)
    v16bf f;
#pragma unroll
    for (int j = 0; j < 8; ++j) f[j] = (__bf16)p0[j];
#pragma unroll
    for (int j = 0; j < 8; ++j) f[8 + j] = (__bf16)p1[j];
    afr[kt] = f;
  }

  // --- sweep N tiles --------------------------------------------------------
  for (int nt = 0; nt < COUT / 16; ++nt) {
    v8f acc = {};
    // B column n = W row (nt*16 + n); lane's K chunks [h*8, h*8+8) and [16+h*8, ...)
    const float* wrow = W + (size_t)(nt * 16 + m) * CIN;
#pragma unroll
    for (int kt = 0; kt < CIN / 32; ++kt) {
      const float* q = wrow + kt * 32 + hh * 8;
      v16bf b;
#pragma unroll
      for (int j = 0; j < 8; ++j) b[j] = (__bf16)q[j];
#pragma unroll
      for (int j = 0; j < 8; ++j) b[8 + j] = (__bf16)q[16 + j];
      acc = __builtin_amdgcn_wmma_f32_16x16x32_bf16(
          /*neg_a=*/false, afr[kt], /*neg_b=*/false, b,
          /*c_mod=*/(short)0, acc, /*reuse_a=*/false, /*reuse_b=*/false);
    }
    const float bv = bias ? bias[nt * 16 + m] : 0.0f;
    float* orow = out + (size_t)row0 * COUT + nt * 16 + m;
#pragma unroll
    for (int i = 0; i < 8; ++i) {                 // D: VGPR i -> row i + 8*hh
      orow[(size_t)(i + 8 * hh) * COUT] = acc[i] + bv;
    }
  }
}

// ---------------------------------------------------------------------------
// init: deg = 1.0 (self loop), batch-norm accumulators = 0
// ---------------------------------------------------------------------------
__global__ void init_kernel(float* __restrict__ deg, float* __restrict__ bnsum,
                            float* __restrict__ bnsum2, int n) {
  int t = blockIdx.x * blockDim.x + threadIdx.x;
  if (t < n) deg[t] = 1.0f;
  if (t < CCH) { bnsum[t] = 0.0f; bnsum2[t] = 0.0f; }
}

// deg[dst] += 1 per edge (in-degree + self loop)
__global__ void deg_accum_kernel(float* __restrict__ deg,
                                 const long long* __restrict__ dst,
                                 long long nE) {
  long long e = (long long)blockIdx.x * blockDim.x + threadIdx.x;
  if (e >= nE) return;
  atomicAdd(&deg[(int)dst[e]], 1.0f);
}

// deg -> 1/sqrt(deg) in place (deg >= 1 always)
__global__ void dinv_kernel(float* __restrict__ deg, int n) {
  int t = blockIdx.x * blockDim.x + threadIdx.x;
  if (t < n) deg[t] = rsqrtf(deg[t]);
}

// ---------------------------------------------------------------------------
// xin[n, 0:128] = node_features ; xin[n, 128:256] = cos(ts*freq + phase)
// ---------------------------------------------------------------------------
__global__ void prep_xin_kernel(float* __restrict__ xin,
                                const float* __restrict__ nf,
                                const float* __restrict__ ts,
                                const float* __restrict__ freq,
                                const float* __restrict__ phase, int n) {
  int t = blockIdx.x * blockDim.x + threadIdx.x;
  if (t >= n * CCH) return;
  int node = t >> 7, c = t & (CCH - 1);
  float* row = xin + (size_t)node * (2 * CCH);
  row[c] = nf[t];
  row[CCH + c] = __cosf(ts[node] * freq[c] + phase[c]);
}

// ---------------------------------------------------------------------------
// agg[n,c] = dinv[n]^2 * h[n,c] + b[c]    (self-loop term + GCN bias)
// ---------------------------------------------------------------------------
__global__ void agg_init_kernel(float* __restrict__ agg,
                                const float* __restrict__ h,
                                const float* __restrict__ dinv,
                                const float* __restrict__ bias, int n) {
  int t = blockIdx.x * blockDim.x + threadIdx.x;       // one float4 each
  if (t >= n * (CCH / 4)) return;
  int node = t >> 5, c = (t & 31) * 4;
  float di = dinv[node];
  float ns = di * di;
  float4 hv = *(const float4*)(h + (size_t)node * CCH + c);
  float4 bv = *(const float4*)(bias + c);
  float4 o;
  o.x = hv.x * ns + bv.x; o.y = hv.y * ns + bv.y;
  o.z = hv.z * ns + bv.z; o.w = hv.w * ns + bv.w;
  *(float4*)(agg + (size_t)node * CCH + c) = o;
}

// agg[dst] += h[src] * dinv[src]*dinv[dst]  (32 lanes per edge, coalesced row)
__global__ __launch_bounds__(256) void agg_edges_kernel(
    float* __restrict__ agg, const float* __restrict__ h,
    const float* __restrict__ dinv, const long long* __restrict__ src,
    const long long* __restrict__ dst, long long nE) {
  long long t = (long long)blockIdx.x * 256 + threadIdx.x;
  long long e = t >> 5;
  if (e >= nE) return;
  int c = (int)(t & 31) * 4;
  int s = (int)src[e], d = (int)dst[e];
  float ns = dinv[s] * dinv[d];
  float4 hv = *(const float4*)(h + (size_t)s * CCH + c);
  float* ap = agg + (size_t)d * CCH + c;
  atomicAdd(ap + 0, hv.x * ns);
  atomicAdd(ap + 1, hv.y * ns);
  atomicAdd(ap + 2, hv.z * ns);
  atomicAdd(ap + 3, hv.w * ns);
}

// ---------------------------------------------------------------------------
// GRU gate math (+ optional skip add, optional relu)
// ---------------------------------------------------------------------------
__device__ __forceinline__ float sigmoidf(float x) {
  return 1.0f / (1.0f + __expf(-x));
}

__global__ void gru_combine_kernel(float* __restrict__ out,
                                   const float* __restrict__ gi,
                                   const float* __restrict__ gh,
                                   const float* __restrict__ hprev,
                                   const float* __restrict__ skip,
                                   int doRelu, int n) {
  int t = blockIdx.x * blockDim.x + threadIdx.x;
  if (t >= n * CCH) return;
  int node = t >> 7, c = t & (CCH - 1);
  size_t g = (size_t)node * (3 * CCH) + c;
  float r = sigmoidf(gi[g] + gh[g]);
  float z = sigmoidf(gi[g + CCH] + gh[g + CCH]);
  float nn = tanhf(gi[g + 2 * CCH] + r * gh[g + 2 * CCH]);
  float v = (1.0f - z) * nn + z * hprev[t];
  if (skip) v += skip[t];
  if (doRelu) v = fmaxf(v, 0.0f);
  out[t] = v;
}

// ---------------------------------------------------------------------------
// BatchNorm (training-mode batch stats, weight=1, bias=0)
// ---------------------------------------------------------------------------
__global__ __launch_bounds__(CCH) void bn_reduce_kernel(
    const float* __restrict__ H, float* __restrict__ bnsum,
    float* __restrict__ bnsum2, int n) {
  int c = threadIdx.x;
  float s = 0.0f, s2 = 0.0f;
  for (int node = blockIdx.x; node < n; node += gridDim.x) {
    float v = H[(size_t)node * CCH + c];
    s += v; s2 += v * v;
  }
  atomicAdd(&bnsum[c], s);
  atomicAdd(&bnsum2[c], s2);
}

__global__ __launch_bounds__(CCH) void bn_stats_kernel(
    const float* __restrict__ bnsum, const float* __restrict__ bnsum2,
    float* __restrict__ mean, float* __restrict__ inv, int n) {
  int c = threadIdx.x;
  float m = bnsum[c] / (float)n;
  float var = bnsum2[c] / (float)n - m * m;
  mean[c] = m;
  inv[c] = rsqrtf(var + BNEPS);
}

__global__ void bn_apply_kernel(float* __restrict__ out,
                                const float* __restrict__ H,
                                const float* __restrict__ mean,
                                const float* __restrict__ inv, int n) {
  int t = blockIdx.x * blockDim.x + threadIdx.x;
  if (t >= n * CCH) return;
  int c = t & (CCH - 1);
  out[t] = (H[t] - mean[c]) * inv[c];
}

// ---------------------------------------------------------------------------
// Host-side orchestration
// ---------------------------------------------------------------------------
static inline int ceil_div(long long a, long long b) { return (int)((a + b - 1) / b); }

extern "C" void kernel_launch(void* const* d_in, const int* in_sizes, int n_in,
                              void* d_out, int out_size, void* d_ws, size_t ws_size,
                              hipStream_t stream) {
  const int N = in_sizes[0] / CCH;                    // 50000
  const long long E = (long long)in_sizes[1] / 2;     // 1.6M

  const float*     nf       = (const float*)d_in[0];
  const long long* eidx     = (const long long*)d_in[1];   // int64 [2, E]
  const float*     x_prev1  = (const float*)d_in[2];
  const float*     x_prev2  = (const float*)d_in[3];
  const float*     ts       = (const float*)d_in[4];
  const float*     freq     = (const float*)d_in[5];
  const float*     phase    = (const float*)d_in[6];
  const float*     merge_W  = (const float*)d_in[7];
  const float*     merge_b  = (const float*)d_in[8];
  const float*     W1       = (const float*)d_in[9];
  const float*     b1       = (const float*)d_in[10];
  const float*     g1_Wih   = (const float*)d_in[11];
  const float*     g1_Whh   = (const float*)d_in[12];
  const float*     g1_bih   = (const float*)d_in[13];
  const float*     g1_bhh   = (const float*)d_in[14];
  const float*     W2       = (const float*)d_in[15];
  const float*     b2       = (const float*)d_in[16];
  const float*     g2_Wih   = (const float*)d_in[17];
  const float*     g2_Whh   = (const float*)d_in[18];
  const float*     g2_bih   = (const float*)d_in[19];
  const float*     g2_bhh   = (const float*)d_in[20];
  const float*     skip_W   = (const float*)d_in[21];
  const float*     skip_b   = (const float*)d_in[22];

  const long long* src = eidx;
  const long long* dst = eidx + E;

  float* H1out = (float*)d_out;                       // [N, C]
  float* H2out = (float*)d_out + (size_t)N * CCH;     // [N, C]

  // ---- workspace layout (floats); all slabs 16B-aligned since 4*N % 16 == 0
  float* ws    = (float*)d_ws;
  size_t Nn    = (size_t)N;
  float* deg   = ws;                       // [N]       degree -> 1/sqrt(deg)
  float* xin   = deg  + Nn;                // [N, 256]  merged input
  float* xbuf  = xin  + Nn * 256;          // [N, 128]  merge output x
  float* hbuf  = xbuf + Nn * 128;          // [N, 128]  per-layer x@W (and skip)
  float* agg   = hbuf + Nn * 128;          // [N, 128]  GCN aggregation / H2raw
  float* gibuf = agg  + Nn * 128;          // [N, 384]
  float* ghbuf = gibuf + Nn * 384;         // [N, 384]
  float* bnsum = ghbuf + Nn * 384;         // [128]
  float* bnsum2 = bnsum + CCH;             // [128]
  float* bnmean = bnsum2 + CCH;            // [128]
  float* bninv  = bnmean + CCH;            // [128]
  (void)ws_size; (void)n_in; (void)out_size;

  const int TPB = 256;
  const int gElemC  = ceil_div((long long)N * CCH, TPB);        // N*C threads
  const int gElemC4 = ceil_div((long long)N * (CCH / 4), TPB);  // N*C/4 threads
  const int gEdge   = ceil_div(E, TPB);
  const int gEdge32 = ceil_div(E * 32, TPB);
  const int gGemm   = ceil_div((long long)(N >> 4), 4);         // 4 waves/block

  // 1) degrees / norms (shared by both GCN layers)
  init_kernel<<<ceil_div(N, TPB), TPB, 0, stream>>>(deg, bnsum, bnsum2, N);
  deg_accum_kernel<<<gEdge, TPB, 0, stream>>>(deg, dst, E);
  dinv_kernel<<<ceil_div(N, TPB), TPB, 0, stream>>>(deg, N);

  // 2) time encode + merge:  x = [nf | cos(ts*f+p)] @ merge_W^T + merge_b
  prep_xin_kernel<<<gElemC, TPB, 0, stream>>>(xin, nf, ts, freq, phase, N);
  gemm_wmma_kernel<256, 128><<<gGemm, 128, 0, stream>>>(xbuf, xin, merge_W, merge_b, N);

  // 3) layer 1: GCNConv + GRU + relu  -> H1 (first output)
  gemm_wmma_kernel<128, 128><<<gGemm, 128, 0, stream>>>(hbuf, xbuf, W1, nullptr, N);
  agg_init_kernel<<<gElemC4, TPB, 0, stream>>>(agg, hbuf, deg, b1, N);
  agg_edges_kernel<<<gEdge32, TPB, 0, stream>>>(agg, hbuf, deg, src, dst, E);
  gemm_wmma_kernel<128, 384><<<gGemm, 128, 0, stream>>>(gibuf, agg, g1_Wih, g1_bih, N);
  gemm_wmma_kernel<128, 384><<<gGemm, 128, 0, stream>>>(ghbuf, x_prev1, g1_Whh, g1_bhh, N);
  gru_combine_kernel<<<gElemC, TPB, 0, stream>>>(H1out, gibuf, ghbuf, x_prev1,
                                                 nullptr, /*relu=*/1, N);

  // 4) layer 2: GCNConv + GRU + skip -> H2raw (in agg)
  gemm_wmma_kernel<128, 128><<<gGemm, 128, 0, stream>>>(hbuf, H1out, W2, nullptr, N);
  agg_init_kernel<<<gElemC4, TPB, 0, stream>>>(agg, hbuf, deg, b2, N);
  agg_edges_kernel<<<gEdge32, TPB, 0, stream>>>(agg, hbuf, deg, src, dst, E);
  gemm_wmma_kernel<128, 384><<<gGemm, 128, 0, stream>>>(gibuf, agg, g2_Wih, g2_bih, N);
  gemm_wmma_kernel<128, 384><<<gGemm, 128, 0, stream>>>(ghbuf, x_prev2, g2_Whh, g2_bhh, N);
  gemm_wmma_kernel<128, 128><<<gGemm, 128, 0, stream>>>(hbuf, xbuf, skip_W, skip_b, N);
  gru_combine_kernel<<<gElemC, TPB, 0, stream>>>(agg, gibuf, ghbuf, x_prev2,
                                                 /*skip=*/hbuf, /*relu=*/0, N);

  // 5) batch norm over N -> H2 (second output)
  bn_reduce_kernel<<<256, CCH, 0, stream>>>(agg, bnsum, bnsum2, N);
  bn_stats_kernel<<<1, CCH, 0, stream>>>(bnsum, bnsum2, bnmean, bninv, N);
  bn_apply_kernel<<<gElemC, TPB, 0, stream>>>(H2out, agg, bnmean, bninv, N);
}
